// LSTM_decomposedGIN_28140625724085
// MI455X (gfx1250) — matrix operand order repaired and checked by reading, not compiled, not run on hardware
//
#include <hip/hip_runtime.h>

// ---------------- types for WMMA ----------------
typedef __attribute__((ext_vector_type(16))) __bf16       v16bf;
typedef __attribute__((ext_vector_type(8)))  float        v8f;
typedef __attribute__((ext_vector_type(4)))  unsigned int u32x4;

struct Frag32B { u32x4 lo, hi; };
static_assert(sizeof(Frag32B) == 32, "frag size");
static_assert(sizeof(v16bf)   == 32, "v16bf size");

#define T_STEPS 215
#define BATCH   2048
#define DIN     36
#define DPAD    64      // K-padded input dim for WMMA (36 -> 64, zeros)
#define HID     128
#define G4      512     // 4*H gate columns
#define BT      16      // batch rows per workgroup (one WMMA M tile)

__device__ __forceinline__ unsigned short f2bf(float f) {
  unsigned int u = __float_as_uint(f);
  unsigned int r = u + 0x7FFFu + ((u >> 16) & 1u);   // round-to-nearest-even
  return (unsigned short)(r >> 16);
}

__device__ __forceinline__ float sigmoidf(float x) {
  return 1.0f / (1.0f + __expf(-x));
}

// B fragment: two 16B chunks 32 bytes apart (K = half*8..+7 and half*8+16..+23)
__device__ __forceinline__ v16bf load_frag(const unsigned short* p) {
  Frag32B f;
  f.lo = *(const u32x4*)(p);
  f.hi = *(const u32x4*)(p + 16);
  return __builtin_bit_cast(v16bf, f);
}
// A fragment: two 16B chunks 32 bytes apart (K = half*8..+7 and half*8+16..+23)
__device__ __forceinline__ v16bf load_frag_split(const unsigned short* p) {
  Frag32B f;
  f.lo = *(const u32x4*)(p);
  f.hi = *(const u32x4*)(p + 16);
  return __builtin_bit_cast(v16bf, f);
}

// ---------------------------------------------------------------------------
// Kernel 0: weight prep. fp32 -> bf16, pad Wx K dim 36->64, fuse biases.
// ---------------------------------------------------------------------------
__global__ __launch_bounds__(256) void lstm_prep_kernel(
    const float* __restrict__ Wx, const float* __restrict__ bx,
    const float* __restrict__ Wh, const float* __restrict__ bh,
    unsigned short* __restrict__ wxb,   // [512][64]
    unsigned short* __restrict__ whb,   // [512][128]
    float* __restrict__ bias)           // [512]
{
  int i = blockIdx.x * 256 + threadIdx.x;
  if (i < G4) bias[i] = bx[i] + bh[i];
  if (i < G4 * DPAD) {
    int n = i >> 6, k = i & 63;
    wxb[i] = (k < DIN) ? f2bf(Wx[n * DIN + k]) : (unsigned short)0;
  }
  if (i < G4 * HID) whb[i] = f2bf(Wh[i]);
}

// ---------------------------------------------------------------------------
// Kernel 1: encoder GEMM + scale + GIN affine chain -> bf16 [T][B][64]
// grid = (B/64, T), block = 256
// ---------------------------------------------------------------------------
__global__ __launch_bounds__(256) void enc_gin_kernel(
    const float* __restrict__ src,   // [T][B][36]
    const float* __restrict__ encW,  // [36][36]
    const float* __restrict__ encb,  // [36]
    const float* __restrict__ gw1, const float* __restrict__ gb1,
    const float* __restrict__ ggamma, const float* __restrict__ gbeta,
    const float* __restrict__ gw2, const float* __restrict__ gb2,
    unsigned short* __restrict__ xg) // [T][B][64] bf16
{
  __shared__ float sW[DIN * DIN];
  __shared__ float sS[64 * DIN];
  __shared__ float sB[DIN];
  __shared__ float sX0[DIN];
  __shared__ float sSum;

  const int tid = threadIdx.x;
  const int t   = blockIdx.y;
  const int b0  = blockIdx.x * 64;

  for (int i = tid; i < DIN * DIN; i += 256) sW[i] = encW[i];
  if (tid < DIN) sB[tid] = encb[tid];
  const float* srcT = src + ((size_t)t * BATCH + b0) * DIN;
  for (int i = tid; i < 64 * DIN; i += 256) sS[i] = srcT[i];
  __syncthreads();

  float xv[9];
#pragma unroll
  for (int r = 0; r < 9; ++r) {             // 64*36 = 2304 = 256*9 outputs
    int idx = tid + 256 * r;
    int bl = idx / DIN, d = idx - bl * DIN;
    float acc = sB[d];
    const float* wrow = &sW[d * DIN];
    const float* srow = &sS[bl * DIN];
#pragma unroll 4
    for (int k = 0; k < DIN; ++k) acc += srow[k] * wrow[k];
    float x = acc * 8.0f;                    // * sqrt(d_model)
    xv[r] = x;
    if (b0 == 0 && bl == 0) sX0[d] = x;      // only r==0, tid<36 hits this
  }
  __syncthreads();
  if (tid == 0) {
    float s = 0.0f;
    if (b0 == 0) {
      for (int d = 0; d < DIN; ++d) s += sX0[d];
    }
    sSum = 2.0f * s;                         // each target gets S twice
  }
  __syncthreads();

  const float W1 = gw1[0], B1 = gb1[0], GM = ggamma[0], BB = gbeta[0];
  const float W2 = gw2[0], B2 = gb2[0];
  const float bn = rsqrtf(1.0f + 1e-5f);     // BN eval: mean 0, var 1

#pragma unroll
  for (int r = 0; r < 9; ++r) {
    int idx = tid + 256 * r;
    int bl = idx / DIN, d = idx - bl * DIN;
    float h = xv[r] + ((b0 == 0 && bl == 0) ? sSum : 0.0f); // GIN agg quirk
    h = h * W1 + B1;
    h = h * bn * GM + BB;
    h = fmaxf(h, 0.0f);
    h = h * W2 + B2;
    h = fmaxf(h, 0.0f);
    xg[((size_t)t * BATCH + b0 + bl) * DPAD + d] = f2bf(h);
  }
  // zero K-padding (cols 36..63): 64*28 = 1792 = 256*7
#pragma unroll
  for (int r = 0; r < 7; ++r) {
    int i = tid + 256 * r;
    int bl = i / (DPAD - DIN), d = DIN + (i - bl * (DPAD - DIN));
    xg[((size_t)t * BATCH + b0 + bl) * DPAD + d] = 0;
  }
}

// ---------------------------------------------------------------------------
// Kernel 2: persistent batched LSTM via WMMA.
// grid = B/16 blocks; block = 256 threads (8 waves). Each block owns 16 batch
// rows for all 215 steps; each wave owns 64 gate columns (4 N-tiles).
// gates[16,512] = x_t[16,64]@WxT + h[16,128]@WhT + bias (bf16 A/B, f32 acc)
// ---------------------------------------------------------------------------
__global__ __launch_bounds__(256) void lstm_kernel(
    const unsigned short* __restrict__ xg,   // [T][B][64] bf16
    const unsigned short* __restrict__ wxb,  // [512][64]  bf16 (N-major)
    const unsigned short* __restrict__ whb,  // [512][128] bf16 (N-major)
    const float* __restrict__ bias,          // [512]
    const int* __restrict__ lengths,         // [B]
    float* __restrict__ pooled)              // [B][128]
{
  __shared__ __align__(16) float          gLDS[BT][G4 + 8];  // gates
  __shared__ __align__(16) unsigned short hLDS[BT][HID];     // h (bf16)

  const int tid  = threadIdx.x;
  const int wave = tid >> 5;
  const int lane = tid & 31;
  const int lrow = lane & 15;
  const int half = lane >> 4;
  const int b0   = blockIdx.x * BT;

  // zero initial hidden state
  for (int i = tid; i < BT * HID; i += 256) (&hLDS[0][0])[i] = 0;

  // per-wave weight row base: lane's N column is fixed for all fragments
  const int nBase = wave * 64 + lrow;
  const unsigned short* wxRow = wxb + (size_t)nBase * DPAD + half * 8;
  const unsigned short* whRow = whb + (size_t)nBase * HID + half * 8;

  float bfrag[4];
#pragma unroll
  for (int nt = 0; nt < 4; ++nt) bfrag[nt] = bias[wave * 64 + nt * 16 + lrow];

  // element-wise ownership: thread -> (batch row eb, 8 h-cols from eh0)
  const int eb  = tid >> 4;
  const int eh0 = (tid & 15) * 8;
  const int mylen = lengths[b0 + eb];
  float creg[8], pool[8];
#pragma unroll
  for (int s = 0; s < 8; ++s) { creg[s] = 0.0f; pool[s] = 0.0f; }

  __syncthreads();

  for (int t = 0; t < T_STEPS; ++t) {
    const unsigned short* xrow =
        xg + ((size_t)t * BATCH + b0 + lrow) * DPAD;
    if (t + 1 < T_STEPS)  // warm L2/WGP$ for next step  -> global_prefetch_b8
      __builtin_prefetch(xg + ((size_t)(t + 1) * BATCH + b0 + lrow) * DPAD, 0, 1);

    // A fragments: x_t (2 K-steps of 32) and h (4 K-steps of 32)
    v16bf ax0 = load_frag_split(xrow + half * 8);
    v16bf ax1 = load_frag_split(xrow + 32 + half * 8);
    v16bf ah[4];
#pragma unroll
    for (int ks = 0; ks < 4; ++ks)
      ah[ks] = load_frag_split(&hLDS[lrow][ks * 32 + half * 8]);

#pragma unroll
    for (int nt = 0; nt < 4; ++nt) {
      const float bv = bfrag[nt];
      v8f acc = {bv, bv, bv, bv, bv, bv, bv, bv};
      const unsigned short* wxp = wxRow + (size_t)(nt * 16) * DPAD;
      acc = __builtin_amdgcn_wmma_f32_16x16x32_bf16(
          false, ax0, false, load_frag(wxp), (short)0, acc, false, false);
      acc = __builtin_amdgcn_wmma_f32_16x16x32_bf16(
          false, ax1, false, load_frag(wxp + 32), (short)0, acc, false, false);
      const unsigned short* whp = whRow + (size_t)(nt * 16) * HID;
#pragma unroll
      for (int ks = 0; ks < 4; ++ks)
        acc = __builtin_amdgcn_wmma_f32_16x16x32_bf16(
            false, ah[ks], false, load_frag(whp + ks * 32), (short)0, acc,
            false, false);
      // scatter accumulator to LDS: C layout M = half*8+v, N = lane&15
      const int col = wave * 64 + nt * 16 + lrow;
#pragma unroll
      for (int v = 0; v < 8; ++v) gLDS[half * 8 + v][col] = acc[v];
    }
    __syncthreads();

    // element-wise LSTM cell + masked pooling + write h back (bf16)
#pragma unroll
    for (int s = 0; s < 8; ++s) {
      const int h = eh0 + s;
      const float gi = gLDS[eb][h];
      const float gf = gLDS[eb][128 + h];
      const float gg = gLDS[eb][256 + h];
      const float go = gLDS[eb][384 + h];
      const float c  = creg[s] * sigmoidf(gf) + sigmoidf(gi) * tanhf(gg);
      const float hv = sigmoidf(go) * tanhf(c);
      creg[s] = c;
      if (t < mylen) pool[s] += hv;
      hLDS[eb][h] = f2bf(hv);
    }
    __syncthreads();
  }

  const float inv = 1.0f / (float)(mylen + 1);
#pragma unroll
  for (int s = 0; s < 8; ++s)
    pooled[(size_t)(b0 + eb) * HID + eh0 + s] = pool[s] * inv;
}

// ---------------------------------------------------------------------------
// Kernel 3: MLP head. one block per batch row.
// ---------------------------------------------------------------------------
#define DMLP 137
__global__ __launch_bounds__(160) void mlp_kernel(
    const float* __restrict__ pooled,  // [B][128]
    const float* __restrict__ stat,    // [B][9]
    const float* __restrict__ W1, const float* __restrict__ b1,
    const float* __restrict__ W2, const float* __restrict__ b2,
    float* __restrict__ out)           // [B][2]
{
  __shared__ float feat[DMLP];
  __shared__ float hid[DMLP];
  const int b = blockIdx.x, tid = threadIdx.x;
  if (tid < HID)        feat[tid] = pooled[(size_t)b * HID + tid];
  else if (tid < DMLP)  feat[tid] = stat[(size_t)b * 9 + (tid - HID)];
  __syncthreads();
  if (tid < DMLP) {
    float a = b1[tid];
    const float* w = W1 + (size_t)tid * DMLP;
#pragma unroll 4
    for (int k = 0; k < DMLP; ++k) a += w[k] * feat[k];
    hid[tid] = fmaxf(a, 0.0f);
  }
  __syncthreads();
  if (tid < 2) {
    float a = b2[tid];
    const float* w = W2 + (size_t)tid * DMLP;
#pragma unroll 4
    for (int k = 0; k < DMLP; ++k) a += w[k] * hid[k];
    out[(size_t)b * 2 + tid] = a;
  }
}

// ---------------------------------------------------------------------------
extern "C" void kernel_launch(void* const* d_in, const int* in_sizes, int n_in,
                              void* d_out, int out_size, void* d_ws,
                              size_t ws_size, hipStream_t stream) {
  (void)in_sizes; (void)n_in; (void)out_size; (void)ws_size;
  const float* src     = (const float*)d_in[0];
  const float* stat    = (const float*)d_in[1];
  /* times d_in[2] unused by reference */
  const int*   lengths = (const int*)d_in[3];
  const float* encW    = (const float*)d_in[4];
  const float* encb    = (const float*)d_in[5];
  const float* gw1     = (const float*)d_in[6];
  const float* gb1     = (const float*)d_in[7];
  const float* ggamma  = (const float*)d_in[8];
  const float* gbeta   = (const float*)d_in[9];
  const float* gw2     = (const float*)d_in[10];
  const float* gb2     = (const float*)d_in[11];
  const float* Wx      = (const float*)d_in[12];
  const float* bx      = (const float*)d_in[13];
  const float* Wh      = (const float*)d_in[14];
  const float* bh      = (const float*)d_in[15];
  const float* W1      = (const float*)d_in[16];
  const float* b1      = (const float*)d_in[17];
  const float* W2      = (const float*)d_in[18];
  const float* b2      = (const float*)d_in[19];
  float* out = (float*)d_out;

  // workspace layout (all 256B-aligned)
  char* ws = (char*)d_ws;
  size_t off = 0;
  unsigned short* xg = (unsigned short*)(ws + off);
  off += (size_t)T_STEPS * BATCH * DPAD * 2;        // 56,360,960
  unsigned short* wxb = (unsigned short*)(ws + off);
  off += (size_t)G4 * DPAD * 2;                     // 65,536
  unsigned short* whb = (unsigned short*)(ws + off);
  off += (size_t)G4 * HID * 2;                      // 131,072
  float* bias = (float*)(ws + off);
  off += (size_t)G4 * 4;                            // 2,048
  float* pooled = (float*)(ws + off);
  off += (size_t)BATCH * HID * 4;                   // 1,048,576

  lstm_prep_kernel<<<(G4 * HID + 255) / 256, 256, 0, stream>>>(
      Wx, bx, Wh, bh, wxb, whb, bias);

  enc_gin_kernel<<<dim3(BATCH / 64, T_STEPS), 256, 0, stream>>>(
      src, encW, encb, gw1, gb1, ggamma, gbeta, gw2, gb2, xg);

  lstm_kernel<<<BATCH / BT, 256, 0, stream>>>(
      xg, wxb, whb, bias, lengths, pooled);

  mlp_kernel<<<BATCH, 160, 0, stream>>>(
      pooled, stat, W1, b1, W2, b2, out);
}
